// TypedCapacityDomainMoEFFN_472446403249
// MI455X (gfx1250) — matrix operand order repaired and checked by reading, not compiled, not run on hardware
//
#include <hip/hip_runtime.h>
#include <cstddef>

// ---------------- problem constants ----------------
constexpr int B  = 32;
constexpr int C  = 16;
constexpr int S  = 128;
constexpr int D  = 256;
constexpr int F  = 1024;
constexpr int E  = 8;
constexpr int CAP = 4;                // ceil(1.0 * 32 / 8)
constexpr int ROWS = B * C * S;       // 65536 token rows
constexpr int TM = 32;                // rows per workgroup
constexpr int FCHUNK = 256;           // F processed per chunk
constexpr int NCHUNK = F / FCHUNK;    // 4
constexpr int KST1 = D / 32;          // 8 k-steps, GEMM1
constexpr int KST2 = FCHUNK / 32;     // 8 k-steps, GEMM2
constexpr int MATS = 17;              // shared + 8 spa + 8 spe
constexpr int MAT_ELEMS = D * F;      // 262144 per matrix

// workspace layout (float indices for the fp32 region)
constexpr size_t PART_BASE = 0;                           // B*C*D
constexpr size_t PART_X    = PART_BASE + (size_t)B*C*D;
constexpr size_t FEATS     = PART_X    + (size_t)B*C*D;   // B*768
constexpr size_t GATES     = FEATS + (size_t)B*3*D;       // 64 floats
constexpr size_t ASSIGN    = GATES + 64;                  // 64 ints
constexpr size_t WPACK_F   = 287232;                      // bf16 region start (32B aligned)

// ---------------- vector types ----------------
typedef __attribute__((ext_vector_type(8)))  float  v8f;
typedef __attribute__((ext_vector_type(8)))  __bf16 v8bf;
typedef __attribute__((ext_vector_type(16))) __bf16 v16bf;

union BF16x16 { v16bf v; v8bf h[2]; };

__device__ __forceinline__ v8f wmma_bf16(v16bf a, v16bf b, v8f c) {
  return __builtin_amdgcn_wmma_f32_16x16x32_bf16(
      false, a, false, b, (short)0, c, false, false);
}

// gelu(v) = v * sigmoid(2k(v + 0.044715 v^3)), k = sqrt(2/pi)
// branch-free: native v_exp_f32 (exp2) + v_rcp_f32; TRANS ops co-execute with WMMA
__device__ __forceinline__ float gelu_fast(float v) {
  const float K2L2E = 1.5957691216057308f * 1.4426950408889634f; // 2k * log2(e)
  float u = (v + 0.044715f * v * v * v) * K2L2E;
  float e = __builtin_amdgcn_exp2f(-u);                 // exp(-2k(...))
  return v * __builtin_amdgcn_rcpf(1.0f + e);           // v * sigmoid
}

// ============ kernel 1: per-(b,c) partial sums over S ============
__global__ void partial_sums_kernel(const float* __restrict__ base,
                                    const float* __restrict__ x,
                                    float* __restrict__ pb,
                                    float* __restrict__ px) {
  int bc = blockIdx.x;
  int d  = threadIdx.x;
  size_t off = (size_t)bc * S * D + d;
  float sb = 0.f, sx = 0.f;
  #pragma unroll 4
  for (int s = 0; s < S; ++s) {
    sb += base[off + (size_t)s * D];
    sx += x[off + (size_t)s * D];
  }
  pb[(size_t)bc * D + d] = sb;
  px[(size_t)bc * D + d] = sx;
}

// ============ kernel 2: router features (B x 3D) ============
__global__ void feats_kernel(const float* __restrict__ pb,
                             const float* __restrict__ px,
                             float* __restrict__ feats) {
  int b = blockIdx.x;
  int d = threadIdx.x;
  float sb = 0.f, sx = 0.f;
  #pragma unroll
  for (int c = 0; c < C; ++c) {
    sb += pb[(size_t)(b * C + c) * D + d];
    sx += px[(size_t)(b * C + c) * D + d];
  }
  const float inv = 1.0f / (float)(C * S);
  float mb = sb * inv, mx = sx * inv;
  feats[(size_t)b * 3 * D + d]         = mb;
  feats[(size_t)b * 3 * D + D + d]     = mx;
  feats[(size_t)b * 3 * D + 2 * D + d] = mx - mb;
}

// ============ kernel 3: router logits + capacity assignment ============
__global__ void router_kernel(const float* __restrict__ feats,
                              const float* __restrict__ spa_rw,
                              const float* __restrict__ spa_rb,
                              const float* __restrict__ spe_rw,
                              const float* __restrict__ spe_rb,
                              float* __restrict__ gates,
                              int* __restrict__ assign) {
  __shared__ float lg[2][B][E];
  int tid = threadIdx.x;
  int i = tid >> 3;
  int e = tid & 7;
  {
    const float* fr = feats + (size_t)i * 3 * D;
    float accA = spa_rb[e], accB = spe_rb[e];
    #pragma unroll 8
    for (int k = 0; k < 3 * D; ++k) {
      float fv = fr[k];
      accA += fv * spa_rw[(size_t)k * E + e];
      accB += fv * spe_rw[(size_t)k * E + e];
    }
    lg[0][i][e] = accA;
    lg[1][i][e] = accB;
  }
  __syncthreads();
  if (tid == 0) {
    for (int bank = 0; bank < 2; ++bank) {
      float (*L)[E] = lg[bank];
      float conf[B];
      for (int s = 0; s < B; ++s) {
        float m = L[s][0];
        for (int q = 1; q < E; ++q) m = fmaxf(m, L[s][q]);
        conf[s] = m;
      }
      int order[B]; bool taken[B];
      for (int s = 0; s < B; ++s) taken[s] = false;
      for (int pos = 0; pos < B; ++pos) {
        int best = -1;
        for (int s = 0; s < B; ++s)
          if (!taken[s] && (best < 0 || conf[s] > conf[best])) best = s;
        taken[best] = true;
        order[pos] = best;
      }
      int counts[E]; for (int q = 0; q < E; ++q) counts[q] = 0;
      int assigned[B];
      for (int pos = 0; pos < B; ++pos) {
        int s = order[pos];
        int rk[E]; bool used[E];
        for (int q = 0; q < E; ++q) used[q] = false;
        for (int p2 = 0; p2 < E; ++p2) {
          int bb = -1;
          for (int q = 0; q < E; ++q)
            if (!used[q] && (bb < 0 || L[s][q] > L[s][bb])) bb = q;
          used[bb] = true;
          rk[p2] = bb;
        }
        int ex = -1;
        for (int p2 = 0; p2 < E; ++p2) {
          int q = rk[p2];
          if (counts[q] < CAP) { ex = q; counts[q]++; break; }
        }
        assigned[s] = ex;
      }
      for (int s = 0; s < B; ++s) {
        float mx = conf[s], sum = 0.f, p[E];
        for (int q = 0; q < E; ++q) { p[q] = expf(L[s][q] - mx); sum += p[q]; }
        int a = assigned[s];
        gates[bank * B + s]  = (a >= 0) ? p[a] / sum : 0.0f;
        assign[bank * B + s] = (a >= 0) ? a : 0;
      }
    }
  }
}

// ============ kernel 4: repack fp32 weights -> bf16 WMMA B-fragments ============
// W1 frag layout: [nt(64)][ks(8)][lane(32)][16]  (n = nt*16+(lane&15), k = ks*32+(lane>>4)*16+e)
// W2 frag layout: [nt(16)][kg(32)][lane(32)][16] (d = nt*16+(lane&15), f = kg*32+(lane>>4)*16+e)
__global__ void repack_kernel(const float* __restrict__ sw1,
                              const float* __restrict__ aw1,
                              const float* __restrict__ ew1,
                              const float* __restrict__ sw2,
                              const float* __restrict__ aw2,
                              const float* __restrict__ ew2,
                              __bf16* __restrict__ wp1,
                              __bf16* __restrict__ wp2) {
  const int CHUNKS_PER_MAT = MAT_ELEMS / 16;  // 16384
  size_t c = (size_t)blockIdx.x * blockDim.x + threadIdx.x;  // chunk id
  int side = (c >= (size_t)MATS * CHUNKS_PER_MAT) ? 1 : 0;
  if (side) c -= (size_t)MATS * CHUNKS_PER_MAT;
  int m = (int)(c >> 14);       // matrix 0..16
  int r = (int)(c & 16383);     // chunk in matrix
  int lane = r & 31;

  const float* src;
  if (m == 0)      src = side ? sw2 : sw1;
  else if (m <= 8) src = (side ? aw2 : aw1) + (size_t)(m - 1) * MAT_ELEMS;
  else             src = (side ? ew2 : ew1) + (size_t)(m - 9) * MAT_ELEMS;

  BF16x16 u;
  if (!side) {
    int ks = (r >> 5) & 7;
    int nt = r >> 8;
    int f  = nt * 16 + (lane & 15);
    int kb = ks * 32 + (lane >> 4) * 16;
    #pragma unroll
    for (int e = 0; e < 16; ++e)
      u.v[e] = (__bf16)src[(size_t)(kb + e) * F + f];
    *(v16bf*)(wp1 + (size_t)m * MAT_ELEMS + (size_t)r * 16) = u.v;
  } else {
    int kg = (r >> 5) & 31;
    int nt = r >> 10;
    int d  = nt * 16 + (lane & 15);
    int fb = kg * 32 + (lane >> 4) * 16;
    #pragma unroll
    for (int e = 0; e < 16; ++e)
      u.v[e] = (__bf16)src[(size_t)(fb + e) * D + d];
    *(v16bf*)(wp2 + (size_t)m * MAT_ELEMS + (size_t)r * 16) = u.v;
  }
}

// ============ kernel 5: fused 3-bank FFN via bf16 WMMA ============
__global__ void __launch_bounds__(256)
moe_ffn_kernel(const float* __restrict__ xg,
               const __bf16* __restrict__ wp1,
               const __bf16* __restrict__ wp2,
               const float* __restrict__ sb1, const float* __restrict__ sb2,
               const float* __restrict__ ab1, const float* __restrict__ ab2,
               const float* __restrict__ eb1, const float* __restrict__ eb2,
               const float* __restrict__ gates,
               const int*   __restrict__ assign,
               float* __restrict__ out) {
  __shared__ __bf16 Xs[TM][D + 8];        // 16.9 KB
  __shared__ __bf16 Hs[TM][FCHUNK + 8];   // 16.9 KB
  __shared__ float  biasSh[256];

  const int tid  = threadIdx.x;
  const int wave = tid >> 5;
  const int lane = tid & 31;
  const int mt     = wave & 1;
  const int ntBase = (wave >> 1) * 4;
  const int half   = lane >> 4;
  const int l15    = lane & 15;
  const int laneOff = lane * 16;           // element offset within fragment
  const int row0 = blockIdx.x * TM;
  const int b = row0 / (C * S);            // uniform per block

  for (int i = tid; i < TM * D; i += 256) {
    int m = i >> 8, k = i & 255;
    Xs[m][k] = (__bf16)xg[(size_t)(row0 + m) * D + k];
  }

  v8f fin[4];
  const v8f vzero = {0.f,0.f,0.f,0.f,0.f,0.f,0.f,0.f};
  #pragma unroll
  for (int t = 0; t < 4; ++t) fin[t] = vzero;

  __syncthreads();

  const __bf16* aRow = &Xs[mt * 16 + l15][0];
  const __bf16* hRow = &Hs[mt * 16 + l15][0];

  for (int bank = 0; bank < 3; ++bank) {
    int midx; const float *b1p, *b2p; float gate;
    if (bank == 0) {
      midx = 0; b1p = sb1; b2p = sb2; gate = 1.0f;
    } else {
      int bi = bank - 1;
      int a  = assign[bi * B + b];
      gate   = gates[bi * B + b];
      midx   = (bi ? 9 : 1) + a;
      b1p = (bi ? eb1 : ab1) + (size_t)a * F;
      b2p = (bi ? eb2 : ab2) + (size_t)a * D;
    }
    const __bf16* w1f = wp1 + (size_t)midx * MAT_ELEMS;
    const __bf16* w2f = wp2 + (size_t)midx * MAT_ELEMS;

    v8f c2[4];
    #pragma unroll
    for (int t = 0; t < 4; ++t) c2[t] = vzero;

    for (int fc = 0; fc < NCHUNK; ++fc) {
      const int f0 = fc * FCHUNK;

      // ---- GEMM1: C1 = X @ W1 chunk (no barriers; weights direct from global) ----
      v8f c1[4];
      #pragma unroll
      for (int t = 0; t < 4; ++t) c1[t] = vzero;
      biasSh[tid] = b1p[f0 + tid];

      #pragma unroll 2
      for (int ks = 0; ks < KST1; ++ks) {
        const int k0 = ks * 32;
        BF16x16 ua;
        ua.h[0] = *(const v8bf*)(aRow + k0 + half * 16);
        ua.h[1] = *(const v8bf*)(aRow + k0 + half * 16 + 8);
        #pragma unroll
        for (int t = 0; t < 4; ++t) {
          int ntg = fc * 16 + ntBase + t;           // global f-tile
          v16bf bv = *(const v16bf*)(w1f + (((size_t)(ntg * 8 + ks)) << 9) + laneOff);
          c1[t] = wmma_bf16(ua.v, bv, c1[t]);
        }
      }

      // ---- epilogue: bias + gelu (branch-free) -> Hs ----
      #pragma unroll
      for (int t = 0; t < 4; ++t) {
        int ncol = (ntBase + t) * 16 + l15;
        float bb = biasSh[ncol];
        #pragma unroll
        for (int r = 0; r < 8; ++r) {
          int mrow = mt * 16 + r + half * 8;
          Hs[mrow][ncol] = (__bf16)gelu_fast(c1[t][r] + bb);
        }
      }
      __syncthreads();

      // ---- GEMM2 partial: out += H_chunk @ W2 rows ----
      #pragma unroll 2
      for (int ks = 0; ks < KST2; ++ks) {
        const int kf0 = ks * 32;
        const int kg = fc * 8 + ks;                 // global k-step over F
        BF16x16 ua;
        ua.h[0] = *(const v8bf*)(hRow + kf0 + half * 16);
        ua.h[1] = *(const v8bf*)(hRow + kf0 + half * 16 + 8);
        #pragma unroll
        for (int t = 0; t < 4; ++t) {
          int nt = ntBase + t;                      // d-tile
          v16bf bv = *(const v16bf*)(w2f + (((size_t)(nt * 32 + kg)) << 9) + laneOff);
          c2[t] = wmma_bf16(ua.v, bv, c2[t]);
        }
      }
      __syncthreads();   // Hs consumed; safe to rewrite next chunk
    } // fc

    // ---- bank accumulate: fin += gate * (c2 + b2) ----
    biasSh[tid] = b2p[tid];
    __syncthreads();
    #pragma unroll
    for (int t = 0; t < 4; ++t) {
      int ncol = (ntBase + t) * 16 + l15;
      float bb = biasSh[ncol];
      #pragma unroll
      for (int r = 0; r < 8; ++r)
        fin[t][r] += gate * (c2[t][r] + bb);
    }
    __syncthreads();
  } // bank

  // ---- store fp32 output ----
  #pragma unroll
  for (int t = 0; t < 4; ++t) {
    int ncol = (ntBase + t) * 16 + l15;
    #pragma unroll
    for (int r = 0; r < 8; ++r) {
      int mrow = row0 + mt * 16 + r + half * 8;
      out[(size_t)mrow * D + ncol] = fin[t][r];
    }
  }
}

// ---------------- launcher ----------------
extern "C" void kernel_launch(void* const* d_in, const int* in_sizes, int n_in,
                              void* d_out, int out_size, void* d_ws, size_t ws_size,
                              hipStream_t stream) {
  const float* baseline  = (const float*)d_in[0];
  const float* x         = (const float*)d_in[1];
  const float* shared_w1 = (const float*)d_in[2];
  const float* shared_b1 = (const float*)d_in[3];
  const float* shared_w2 = (const float*)d_in[4];
  const float* shared_b2 = (const float*)d_in[5];
  const float* spa_rw    = (const float*)d_in[6];
  const float* spa_rb    = (const float*)d_in[7];
  const float* spe_rw    = (const float*)d_in[8];
  const float* spe_rb    = (const float*)d_in[9];
  const float* spa_w1    = (const float*)d_in[10];
  const float* spa_b1    = (const float*)d_in[11];
  const float* spa_w2    = (const float*)d_in[12];
  const float* spa_b2    = (const float*)d_in[13];
  const float* spe_w1    = (const float*)d_in[14];
  const float* spe_b1    = (const float*)d_in[15];
  const float* spe_w2    = (const float*)d_in[16];
  const float* spe_b2    = (const float*)d_in[17];

  float* ws  = (float*)d_ws;
  float* out = (float*)d_out;
  __bf16* wp1 = (__bf16*)(ws + WPACK_F);
  __bf16* wp2 = wp1 + (size_t)MATS * MAT_ELEMS;

  partial_sums_kernel<<<B * C, 256, 0, stream>>>(baseline, x,
                                                 ws + PART_BASE, ws + PART_X);
  feats_kernel<<<B, 256, 0, stream>>>(ws + PART_BASE, ws + PART_X, ws + FEATS);
  router_kernel<<<1, 256, 0, stream>>>(ws + FEATS, spa_rw, spa_rb, spe_rw, spe_rb,
                                       ws + GATES, (int*)(ws + ASSIGN));
  // 2 * 17 * 16384 chunks / 256 threads = 2176 blocks
  repack_kernel<<<2176, 256, 0, stream>>>(shared_w1, spa_w1, spe_w1,
                                          shared_w2, spa_w2, spe_w2, wp1, wp2);
  moe_ffn_kernel<<<ROWS / TM, 256, 0, stream>>>(
      x, wp1, wp2,
      shared_b1, shared_b2, spa_b1, spa_b2, spe_b1, spe_b2,
      ws + GATES, (const int*)(ws + ASSIGN), out);
}
